// CrossAttention_88888643158567
// MI455X (gfx1250) — matrix operand order repaired and checked by reading, not compiled, not run on hardware
//
#include <hip/hip_runtime.h>

#define DIMC 1024
#define SEQ  2048
#define NH   16
#define HD   64
#define NBATCH 2
#define NTOK (NBATCH * SEQ)
#define ATTN_SCALE 0.125f   // 1/sqrt(64)

typedef __bf16 bf16;
typedef __attribute__((ext_vector_type(16))) __bf16 v16bf;
typedef __attribute__((ext_vector_type(8)))  __bf16 bf16x8;
typedef __attribute__((ext_vector_type(8)))  float  v8f;
typedef __attribute__((ext_vector_type(4)))  float  v4f;

static __device__ __forceinline__ v8f wmma_bf16(v16bf a, v16bf b, v8f c) {
  // (neg_a, A, neg_b, B, c_mod, C, reuse_a, reuse_b)
  return __builtin_amdgcn_wmma_f32_16x16x32_bf16(false, a, false, b, (short)0, c,
                                                 false, false);
}

// A-matrix fragment (16x32 bf16): caller passes &row[k0 + half*16];
// 16 contiguous K values per lane-half.
static __device__ __forceinline__ v16bf load_afrag(const bf16* p) {
  bf16x8 lo = *(const bf16x8*)(p);
  bf16x8 hi = *(const bf16x8*)(p + 8);
  v16bf r;
#pragma unroll
  for (int i = 0; i < 8; ++i) { r[i] = lo[i]; r[i + 8] = hi[i]; }
  return r;
}

// B-matrix fragment (32x16 bf16): caller passes &col[k0 + half*16];
// 16 contiguous K values per lane-half.
static __device__ __forceinline__ v16bf load_bfrag(const bf16* p) {
  bf16x8 lo = *(const bf16x8*)(p);
  bf16x8 hi = *(const bf16x8*)(p + 8);
  v16bf r;
#pragma unroll
  for (int i = 0; i < 8; ++i) { r[i] = lo[i]; r[i + 8] = hi[i]; }
  return r;
}

// ---------------------------------------------------------------- convert
__global__ void cvt_f32_bf16(const float* __restrict__ s, bf16* __restrict__ d,
                             int n) {
  int i = blockIdx.x * blockDim.x + threadIdx.x;
  int stride = gridDim.x * blockDim.x;
  for (; i < n; i += stride) d[i] = (bf16)s[i];
}

// ------------------------------------------------- Q/K projection (A = W)
// 2x2 register-blocked: each wave computes a 32(outdim) x 32(token) macro-tile.
// C[outdim, token] = sum_k W[outdim,k] * X[token,k]; store row-major bf16.
__global__ void __launch_bounds__(128)
proj_qk_kernel(const bf16* __restrict__ X, const bf16* __restrict__ Wq,
               const bf16* __restrict__ Wk, bf16* __restrict__ Qo,
               bf16* __restrict__ Ko) {
  const bf16* W = blockIdx.z ? Wk : Wq;
  bf16* O = blockIdx.z ? Ko : Qo;
  const int wave = threadIdx.x >> 5, lane = threadIdx.x & 31;
  const int hf = lane >> 4, lm = lane & 15;
  const int mBase = (blockIdx.x * 4 + wave) * 32;  // out-dim macro-tile
  const int nBase = blockIdx.y * 32;               // token macro-tile
  const bf16* wrow0 = W + (size_t)(mBase + lm) * DIMC;
  const bf16* wrow1 = W + (size_t)(mBase + 16 + lm) * DIMC;
  const bf16* xrow0 = X + (size_t)(nBase + lm) * DIMC;
  const bf16* xrow1 = X + (size_t)(nBase + 16 + lm) * DIMC;
  v8f acc00 = {}, acc01 = {}, acc10 = {}, acc11 = {};
  for (int k0 = 0; k0 < DIMC; k0 += 32) {
    const v16bf a0 = load_afrag(wrow0 + k0 + hf * 16);
    const v16bf a1 = load_afrag(wrow1 + k0 + hf * 16);
    const v16bf b0 = load_bfrag(xrow0 + k0 + hf * 16);
    const v16bf b1 = load_bfrag(xrow1 + k0 + hf * 16);
    acc00 = wmma_bf16(a0, b0, acc00);
    acc01 = wmma_bf16(a0, b1, acc01);
    acc10 = wmma_bf16(a1, b0, acc10);
    acc11 = wmma_bf16(a1, b1, acc11);
  }
  // lane holds token nBase+ni*16+lm, out-dims mBase+mi*16+hf*8..+7 contiguous
  const v8f* accs[4] = {&acc00, &acc01, &acc10, &acc11};
#pragma unroll
  for (int mi = 0; mi < 2; ++mi)
#pragma unroll
    for (int ni = 0; ni < 2; ++ni) {
      const v8f a = *accs[mi * 2 + ni];
      bf16x8 r;
#pragma unroll
      for (int i = 0; i < 8; ++i) r[i] = (bf16)a[i];
      *(bf16x8*)(O + (size_t)(nBase + ni * 16 + lm) * DIMC + mBase + mi * 16 +
                 hf * 8) = r;
    }
}

// ------------------------------------------------- V projection (A = X)
// 2x2 register-blocked: 32(token) x 32(outdim) macro-tile per wave.
// C[token, outdim]; store transposed vT[b][h*64+d][n] for contiguous PV loads.
__global__ void __launch_bounds__(128)
proj_v_kernel(const bf16* __restrict__ X, const bf16* __restrict__ Wv,
              bf16* __restrict__ VT) {
  const int wave = threadIdx.x >> 5, lane = threadIdx.x & 31;
  const int hf = lane >> 4, lm = lane & 15;
  const int nBase = (blockIdx.x * 4 + wave) * 32;  // out-dim macro-tile
  const int mBase = blockIdx.y * 32;               // token macro-tile
  const bf16* xrow0 = X + (size_t)(mBase + lm) * DIMC;
  const bf16* xrow1 = X + (size_t)(mBase + 16 + lm) * DIMC;
  const bf16* wrow0 = Wv + (size_t)(nBase + lm) * DIMC;
  const bf16* wrow1 = Wv + (size_t)(nBase + 16 + lm) * DIMC;
  v8f acc00 = {}, acc01 = {}, acc10 = {}, acc11 = {};
  for (int k0 = 0; k0 < DIMC; k0 += 32) {
    const v16bf a0 = load_afrag(xrow0 + k0 + hf * 16);
    const v16bf a1 = load_afrag(xrow1 + k0 + hf * 16);
    const v16bf b0 = load_bfrag(wrow0 + k0 + hf * 16);
    const v16bf b1 = load_bfrag(wrow1 + k0 + hf * 16);
    acc00 = wmma_bf16(a0, b0, acc00);
    acc01 = wmma_bf16(a0, b1, acc01);
    acc10 = wmma_bf16(a1, b0, acc10);
    acc11 = wmma_bf16(a1, b1, acc11);
  }
  const v8f* accs[4] = {&acc00, &acc01, &acc10, &acc11};
#pragma unroll
  for (int mi = 0; mi < 2; ++mi) {
    const int t0 = mBase + mi * 16 + hf * 8;  // global token of acc element 0
    const int bb = t0 >> 11, n0 = t0 & (SEQ - 1);
#pragma unroll
    for (int ni = 0; ni < 2; ++ni) {
      const int outd = nBase + ni * 16 + lm;  // == h*64 + d
      const v8f a = *accs[mi * 2 + ni];
      bf16x8 r;
#pragma unroll
      for (int i = 0; i < 8; ++i) r[i] = (bf16)a[i];
      *(bf16x8*)(VT + ((size_t)(bb * DIMC + outd)) * SEQ + n0) = r;
    }
  }
}

// ---------------------------------------------------- flash attention
__global__ void __launch_bounds__(128)
attn_kernel(const bf16* __restrict__ Q, const bf16* __restrict__ Km,
            const bf16* __restrict__ VT, bf16* __restrict__ CTX) {
  __shared__ bf16 pbuf[4][16][40];  // per-wave P tile, padded stride
  const int wave = threadIdx.x >> 5, lane = threadIdx.x & 31;
  const int hf = lane >> 4, lm = lane & 15;
  const int qTile = blockIdx.x * 4 + wave;  // 0..127
  const int bh = blockIdx.y;                // 0..31
  const int bb = bh >> 4, h = bh & 15;
  const int qBase = qTile * 16;
  const size_t tokBase = (size_t)bb * SEQ;

  const bf16* qrow = Q + (tokBase + qBase + lm) * DIMC + h * HD;
  const v16bf aq0 = load_afrag(qrow + 0 + hf * 16);    // d = 0..31
  const v16bf aq1 = load_afrag(qrow + 32 + hf * 16);   // d = 32..63
  const bf16* vbase = VT + ((size_t)(bb * DIMC + h * HD)) * SEQ;

  float mrun[8], lrun[8];
  v8f o0 = {}, o1 = {}, o2 = {}, o3 = {};
#pragma unroll
  for (int i = 0; i < 8; ++i) { mrun[i] = -3.0e38f; lrun[i] = 0.0f; }

  const int kEnd = qBase + 16;  // causal: keys <= max query in tile
  for (int kb = 0; kb < kEnd; kb += 32) {
    const bf16* krow0 = Km + (tokBase + kb + lm) * DIMC + h * HD;
    const bf16* krow1 = Km + (tokBase + kb + 16 + lm) * DIMC + h * HD;
    v8f s0 = {}, s1 = {};
    s0 = wmma_bf16(aq0, load_bfrag(krow0 + 0 + hf * 16), s0);
    s0 = wmma_bf16(aq1, load_bfrag(krow0 + 32 + hf * 16), s0);
    s1 = wmma_bf16(aq0, load_bfrag(krow1 + 0 + hf * 16), s1);
    s1 = wmma_bf16(aq1, load_bfrag(krow1 + 32 + hf * 16), s1);

    float alpha[8];
#pragma unroll
    for (int i = 0; i < 8; ++i) {
      const int qi = qBase + hf * 8 + i;
      float v0 = (kb + lm <= qi) ? s0[i] * ATTN_SCALE : -__builtin_inff();
      float v1 = (kb + 16 + lm <= qi) ? s1[i] * ATTN_SCALE : -__builtin_inff();
      float mx = fmaxf(v0, v1);  // row lives in one 16-lane half
      mx = fmaxf(mx, __shfl_xor(mx, 1));
      mx = fmaxf(mx, __shfl_xor(mx, 2));
      mx = fmaxf(mx, __shfl_xor(mx, 4));
      mx = fmaxf(mx, __shfl_xor(mx, 8));
      const float mnew = fmaxf(mrun[i], mx);
      alpha[i] = __expf(mrun[i] - mnew);
      const float p0 = __expf(v0 - mnew);
      const float p1 = __expf(v1 - mnew);
      float rs = p0 + p1;
      rs += __shfl_xor(rs, 1);
      rs += __shfl_xor(rs, 2);
      rs += __shfl_xor(rs, 4);
      rs += __shfl_xor(rs, 8);
      lrun[i] = lrun[i] * alpha[i] + rs;
      mrun[i] = mnew;
      pbuf[wave][hf * 8 + i][lm] = (bf16)p0;
      pbuf[wave][hf * 8 + i][16 + lm] = (bf16)p1;
    }
    // wave-synchronous LDS round-trip: C-layout -> A-layout
    asm volatile("s_wait_dscnt 0x0" ::: "memory");
    const v16bf pa = load_afrag(&pbuf[wave][lm][hf * 16]);
#pragma unroll
    for (int i = 0; i < 8; ++i) {
      o0[i] *= alpha[i]; o1[i] *= alpha[i];
      o2[i] *= alpha[i]; o3[i] *= alpha[i];
    }
    o0 = wmma_bf16(pa, load_bfrag(vbase + (size_t)(0 + lm) * SEQ + kb + hf * 16), o0);
    o1 = wmma_bf16(pa, load_bfrag(vbase + (size_t)(16 + lm) * SEQ + kb + hf * 16), o1);
    o2 = wmma_bf16(pa, load_bfrag(vbase + (size_t)(32 + lm) * SEQ + kb + hf * 16), o2);
    o3 = wmma_bf16(pa, load_bfrag(vbase + (size_t)(48 + lm) * SEQ + kb + hf * 16), o3);
  }

#pragma unroll
  for (int i = 0; i < 8; ++i) {
    const float inv = 1.0f / lrun[i];
    bf16* cp = CTX + (tokBase + qBase + hf * 8 + i) * DIMC + h * HD + lm;
    cp[0]  = (bf16)(o0[i] * inv);
    cp[16] = (bf16)(o1[i] * inv);
    cp[32] = (bf16)(o2[i] * inv);
    cp[48] = (bf16)(o3[i] * inv);
  }
}

// ------------------------------------------------- output projection + bias
// 2x2 register-blocked like proj_qk; f32 output with bias.
__global__ void __launch_bounds__(128)
proj_out_kernel(const bf16* __restrict__ CTX, const bf16* __restrict__ Wp,
                const float* __restrict__ bp, float* __restrict__ OUT) {
  const int wave = threadIdx.x >> 5, lane = threadIdx.x & 31;
  const int hf = lane >> 4, lm = lane & 15;
  const int mBase = (blockIdx.x * 4 + wave) * 32;  // out-dim macro-tile
  const int nBase = blockIdx.y * 32;               // token macro-tile
  const bf16* wrow0 = Wp + (size_t)(mBase + lm) * DIMC;
  const bf16* wrow1 = Wp + (size_t)(mBase + 16 + lm) * DIMC;
  const bf16* crow0 = CTX + (size_t)(nBase + lm) * DIMC;
  const bf16* crow1 = CTX + (size_t)(nBase + 16 + lm) * DIMC;
  v8f acc00 = {}, acc01 = {}, acc10 = {}, acc11 = {};
  for (int k0 = 0; k0 < DIMC; k0 += 32) {
    const v16bf a0 = load_afrag(wrow0 + k0 + hf * 16);
    const v16bf a1 = load_afrag(wrow1 + k0 + hf * 16);
    const v16bf b0 = load_bfrag(crow0 + k0 + hf * 16);
    const v16bf b1 = load_bfrag(crow1 + k0 + hf * 16);
    acc00 = wmma_bf16(a0, b0, acc00);
    acc01 = wmma_bf16(a0, b1, acc01);
    acc10 = wmma_bf16(a1, b0, acc10);
    acc11 = wmma_bf16(a1, b1, acc11);
  }
  const v8f* accs[4] = {&acc00, &acc01, &acc10, &acc11};
#pragma unroll
  for (int mi = 0; mi < 2; ++mi) {
    const float* bptr = bp + mBase + mi * 16 + hf * 8;
    const v4f b0 = *(const v4f*)bptr;
    const v4f b1 = *(const v4f*)(bptr + 4);
#pragma unroll
    for (int ni = 0; ni < 2; ++ni) {
      const v8f a = *accs[mi * 2 + ni];
      v4f r0, r1;
#pragma unroll
      for (int i = 0; i < 4; ++i) {
        r0[i] = a[i] + b0[i];
        r1[i] = a[i + 4] + b1[i];
      }
      float* op = OUT + (size_t)(nBase + ni * 16 + lm) * DIMC + mBase +
                  mi * 16 + hf * 8;
      *(v4f*)op = r0;
      *(v4f*)(op + 4) = r1;
    }
  }
}

// ------------------------------------------------------------------ launch
extern "C" void kernel_launch(void* const* d_in, const int* in_sizes, int n_in,
                              void* d_out, int out_size, void* d_ws,
                              size_t ws_size, hipStream_t stream) {
  (void)in_sizes; (void)n_in; (void)out_size; (void)ws_size;
  const float* x  = (const float*)d_in[0];
  // d_in[1] = y, unused (cross_attn_flag == False)
  const float* Wq = (const float*)d_in[2];
  const float* Wk = (const float*)d_in[3];
  const float* Wv = (const float*)d_in[4];
  const float* Wp = (const float*)d_in[5];
  const float* bp = (const float*)d_in[6];

  char* ws = (char*)d_ws;                       // 48 MiB layout
  bf16* xbf  = (bf16*)(ws + (size_t)0);         // 8 MiB
  bf16* qbf  = (bf16*)(ws + ((size_t)8  << 20));
  bf16* kbf  = (bf16*)(ws + ((size_t)16 << 20));
  bf16* vtbf = (bf16*)(ws + ((size_t)24 << 20));
  bf16* cbf  = (bf16*)(ws + ((size_t)32 << 20));
  bf16* wqbf = (bf16*)(ws + ((size_t)40 << 20));
  bf16* wkbf = (bf16*)(ws + ((size_t)42 << 20));
  bf16* wvbf = (bf16*)(ws + ((size_t)44 << 20));
  bf16* wpbf = (bf16*)(ws + ((size_t)46 << 20));

  cvt_f32_bf16<<<dim3(1024), dim3(256), 0, stream>>>(x, xbf, NTOK * DIMC);
  cvt_f32_bf16<<<dim3(256), dim3(256), 0, stream>>>(Wq, wqbf, DIMC * DIMC);
  cvt_f32_bf16<<<dim3(256), dim3(256), 0, stream>>>(Wk, wkbf, DIMC * DIMC);
  cvt_f32_bf16<<<dim3(256), dim3(256), 0, stream>>>(Wv, wvbf, DIMC * DIMC);
  cvt_f32_bf16<<<dim3(256), dim3(256), 0, stream>>>(Wp, wpbf, DIMC * DIMC);

  // 32x32 macro-tiles: (1024/32)/4 = 8 block-x, 4096/32 = 128 block-y
  proj_qk_kernel<<<dim3(8, 128, 2), dim3(128), 0, stream>>>(xbf, wqbf, wkbf,
                                                            qbf, kbf);
  proj_v_kernel<<<dim3(8, 128, 1), dim3(128), 0, stream>>>(xbf, wvbf, vtbf);
  attn_kernel<<<dim3(32, 32, 1), dim3(128), 0, stream>>>(qbf, kbf, vtbf, cbf);
  proj_out_kernel<<<dim3(8, 128, 1), dim3(128), 0, stream>>>(cbf, wpbf, bp,
                                                             (float*)d_out);
}
